// EvolveGCNH_25786983645428
// MI455X (gfx1250) — compile-verified, matrix operand order checked by reading, not run on hardware
//
#include <hip/hip_runtime.h>
#include <hip/hip_fp16.h>
#include <cstdint>

// Problem constants (match reference)
#define T_STEPS 6
#define N_NODES 50000
#define E_EDGES 1600000
#define F0_DIM  166
#define H_DIM   76
#define CH_DIM  510
#define C_DIM   2
#define RSLOPE  (11.0f / 48.0f)

typedef __attribute__((ext_vector_type(16))) _Float16 v16h;
typedef __attribute__((ext_vector_type(8)))  _Float16 v8h;
typedef __attribute__((ext_vector_type(8)))  float    v8f;

// ---------------------------------------------------------------- utilities

__global__ void fill_zero_kernel(uint32_t* __restrict__ p, long long n) {
    long long i = (long long)blockIdx.x * blockDim.x + threadIdx.x;
    if (i < n) p[i] = 0u;
}

__device__ __forceinline__ float wave_reduce_add(float s) {
#pragma unroll
    for (int off = 16; off > 0; off >>= 1) s += __shfl_down(s, off, 32);
    return s;
}

// ------------------------------------------------------------- degree stage

__global__ void deg_count_kernel(const int* __restrict__ s, const int* __restrict__ d,
                                 int* __restrict__ degO, int* __restrict__ degI, int E) {
    int e = blockIdx.x * blockDim.x + threadIdx.x;
    if (e < E) {
        atomicAdd(&degO[s[e]], 1);
        atomicAdd(&degI[d[e]], 1);
    }
}

__global__ void deg_finalize_kernel(const int* __restrict__ degO, const int* __restrict__ degI,
                                    float* __restrict__ rsO, float* __restrict__ rsI, int n) {
    int i = blockIdx.x * blockDim.x + threadIdx.x;
    if (i < n) {
        rsO[i] = rsqrtf(fmaxf((float)degO[i], 1.0f));
        rsI[i] = rsqrtf(fmaxf((float)degI[i], 1.0f));
    }
}

// ------------------------------------------------------------ top-k pooling

// Wave-per-node dot product: lanes stride K (coalesced), shuffle reduce.
__global__ void scores_kernel(const float* __restrict__ X, const float* __restrict__ scorer,
                              int F, int n, float* __restrict__ out) {
    int wave = (blockIdx.x * blockDim.x + threadIdx.x) >> 5;
    int lane = threadIdx.x & 31;
    if (wave >= n) return;
    const float* row = X + (size_t)wave * F;
    float s = 0.0f;
    for (int k = lane; k < F; k += 32) s += row[k] * scorer[k];
    s = wave_reduce_add(s);
    if (lane == 0) out[wave] = s;
}

// Single-block iterative argmax: matches jax.lax.top_k ordering
// (descending values, ties broken by smaller index). Destroys `scores`.
__global__ void topk_kernel(float* __restrict__ scores, int n,
                            int* __restrict__ idx, float* __restrict__ vals) {
    __shared__ float sv[1024];
    __shared__ int   si[1024];
    int tid = threadIdx.x;
    for (int h = 0; h < H_DIM; ++h) {
        float best = -INFINITY;
        int   bi   = 0x7fffffff;
        for (int i = tid; i < n; i += 1024) {
            float v = scores[i];
            if (v > best || (v == best && i < bi)) { best = v; bi = i; }
        }
        sv[tid] = best; si[tid] = bi;
        __syncthreads();
        for (int off = 512; off > 0; off >>= 1) {
            if (tid < off) {
                float v = sv[tid + off]; int j = si[tid + off];
                if (v > sv[tid] || (v == sv[tid] && j < si[tid])) { sv[tid] = v; si[tid] = j; }
            }
            __syncthreads();
        }
        if (tid == 0) {
            idx[h] = si[0];
            vals[h] = sv[0];
            scores[si[0]] = -INFINITY;
        }
        __syncthreads();
    }
}

// x_tilde[f,h] = X[idx[h], f] * tanh(vals[h] / max(||scorer||, 1e-6))
__global__ void gather_xtilde_kernel(const float* __restrict__ X, int F,
                                     const float* __restrict__ scorer,
                                     const int* __restrict__ idx,
                                     const float* __restrict__ vals,
                                     float* __restrict__ xt) {
    __shared__ float nrm;
    if (threadIdx.x == 0) {
        float s = 0.0f;
        for (int k = 0; k < F; ++k) s += scorer[k] * scorer[k];
        nrm = fmaxf(sqrtf(s), 1e-6f);
    }
    __syncthreads();
    int t = blockIdx.x * blockDim.x + threadIdx.x;
    if (t < F * H_DIM) {
        int f = t / H_DIM, h = t - f * H_DIM;
        xt[t] = X[(size_t)idx[h] * F + f] * tanhf(vals[h] / nrm);
    }
}

// ------------------------------------------------------------- matrix GRU

__device__ __forceinline__ float sigmoidf(float x) { return 1.0f / (1.0f + __expf(-x)); }

__global__ void gru_ur_kernel(const float* __restrict__ uW, const float* __restrict__ uU,
                              const float* __restrict__ ub,
                              const float* __restrict__ rW, const float* __restrict__ rU,
                              const float* __restrict__ rb,
                              const float* __restrict__ x, const float* __restrict__ Q,
                              float* __restrict__ u, float* __restrict__ r, int F) {
    int t = blockIdx.x * blockDim.x + threadIdx.x;
    if (t >= F * H_DIM) return;
    int i = t / H_DIM, j = t - i * H_DIM;
    float su = ub[t], sr = rb[t];
    for (int k = 0; k < F; ++k) {
        float xk = x[k * H_DIM + j], qk = Q[k * H_DIM + j];
        su += uW[i * F + k] * xk + uU[i * F + k] * qk;
        sr += rW[i * F + k] * xk + rU[i * F + k] * qk;
    }
    u[t] = sigmoidf(su);
    r[t] = sigmoidf(sr);
}

__global__ void gru_rq_kernel(const float* __restrict__ r, const float* __restrict__ Q,
                              float* __restrict__ rq, int n) {
    int t = blockIdx.x * blockDim.x + threadIdx.x;
    if (t < n) rq[t] = r[t] * Q[t];
}

__global__ void gru_h_kernel(const float* __restrict__ hW, const float* __restrict__ hU,
                             const float* __restrict__ hb,
                             const float* __restrict__ x, const float* __restrict__ rq,
                             const float* __restrict__ u, float* __restrict__ Q, int F) {
    int t = blockIdx.x * blockDim.x + threadIdx.x;
    if (t >= F * H_DIM) return;
    int i = t / H_DIM, j = t - i * H_DIM;
    float s = hb[t];
    for (int k = 0; k < F; ++k)
        s += hW[i * F + k] * x[k * H_DIM + j] + hU[i * F + k] * rq[k * H_DIM + j];
    float hc = tanhf(s);
    float uv = u[t];
    Q[t] = (1.0f - uv) * Q[t] + uv * hc;
}

// ----------------------------------------------------- GEMM staging (f16)

// Ah[M x Kp] = f16( A[m, k] * rowScale[m] ), zero-padded in k.
__global__ void pack_a_f16_kernel(const float* __restrict__ A, int lda,
                                  const float* __restrict__ rowScale,
                                  _Float16* __restrict__ Ah,
                                  int M, int K, int Kp) {
    long long t = (long long)blockIdx.x * blockDim.x + threadIdx.x;
    if (t >= (long long)M * Kp) return;
    int row = (int)(t / Kp), k = (int)(t - (long long)row * Kp);
    float v = 0.0f;
    if (k < K) {
        v = A[(size_t)row * lda + k];
        if (rowScale) v *= rowScale[row];
    }
    Ah[t] = (_Float16)v;
}

// Bt[Np x Kp] = f16( B[k, n] ) transposed, zero-padded.
__global__ void pack_bT_f16_kernel(const float* __restrict__ B, int ldb,
                                   _Float16* __restrict__ Bt,
                                   int K, int Nc, int Kp, int Np) {
    long long t = (long long)blockIdx.x * blockDim.x + threadIdx.x;
    if (t >= (long long)Np * Kp) return;
    int n = (int)(t / Kp), k = (int)(t - (long long)n * Kp);
    float v = (n < Nc && k < K) ? B[(size_t)k * ldb + n] : 0.0f;
    Bt[t] = (_Float16)v;
}

// ----------------------------------------------------- WMMA f32<-f16 GEMM
// C[M x Nc] = Ah @ Bt^T (+ bias). One wave computes a 16 x (16*NT) strip:
// the A fragment is loaded once per K-chunk and reused for NT WMMAs.
// Fragments are two global_load_b128 each (padded buffers: no bounds checks,
// no divergence -> EXEC all-ones throughout).
// Layouts per CDNA5 ISA 7.12.2 (wave32): A lane = row (lane&15),
// K pattern kHalf + {0..7, 16..23}; Bt lane = column; C/D n=lane&15,
// m = vgpr + 8*(lane>=16).

template <int NT>
__global__ __launch_bounds__(256)
void wmma_gemm_f16_kernel(const _Float16* __restrict__ Ah, int Kp,
                          const _Float16* __restrict__ Bt,
                          const float* __restrict__ bias,
                          float* __restrict__ C, int ldc,
                          int M, int Nc, int tilesM, int tilesNG) {
    int lane = threadIdx.x & 31;
    int wave = (blockIdx.x * blockDim.x + threadIdx.x) >> 5;
    if (wave >= tilesM * tilesNG) return;   // wave-uniform: EXEC stays all-ones

    int tm = wave / tilesNG, tg = wave - tm * tilesNG;
    int row0 = tm << 4;
    int col0 = tg * (NT << 4);
    int lanelo = lane & 15;
    int kHalf = (lane >> 4) << 3;           // 0 or 8

    const _Float16* aptr = Ah + (size_t)(row0 + lanelo) * Kp + kHalf;
    const _Float16* bptr = Bt + (size_t)(col0 + lanelo) * Kp + kHalf;

    v8f acc[NT] = {};
    for (int k0 = 0; k0 < Kp; k0 += 32) {
        v8h alo = *(const v8h*)(aptr + k0);
        v8h ahi = *(const v8h*)(aptr + k0 + 16);
        v16h a = __builtin_shufflevector(alo, ahi, 0, 1, 2, 3, 4, 5, 6, 7,
                                         8, 9, 10, 11, 12, 13, 14, 15);
#pragma unroll
        for (int nt = 0; nt < NT; ++nt) {
            const _Float16* bp = bptr + (size_t)nt * 16 * Kp + k0;
            v8h blo = *(const v8h*)bp;
            v8h bhi = *(const v8h*)(bp + 16);
            v16h b = __builtin_shufflevector(blo, bhi, 0, 1, 2, 3, 4, 5, 6, 7,
                                             8, 9, 10, 11, 12, 13, 14, 15);
            acc[nt] = __builtin_amdgcn_wmma_f32_16x16x32_f16(
                false, a, false, b, (short)0, acc[nt], false, false);
        }
    }

#pragma unroll
    for (int nt = 0; nt < NT; ++nt) {
        int n = col0 + (nt << 4) + lanelo;
        if (n >= Nc) break;                 // whole-wave uniform? no: lane-varying tail, plain predication is fine here (no WMMA after)
        float bv = bias ? bias[n] : 0.0f;
#pragma unroll
        for (int v = 0; v < 8; ++v) {
            int m = row0 + v + ((lane >> 4) << 3);
            if (m < M) C[(size_t)m * ldc + n] = acc[nt][v] + bv;
        }
    }
}

// --------------------------------------------------------- graph conv rest

__global__ void edge_scatter_kernel(const float* __restrict__ h,
                                    const int* __restrict__ s, const int* __restrict__ d,
                                    float* __restrict__ agg, int E, int Hc) {
    long long t = (long long)blockIdx.x * blockDim.x + threadIdx.x;
    if (t < (long long)E * Hc) {
        int e = (int)(t / Hc);
        int f = (int)(t - (long long)e * Hc);
        atomicAdd(&agg[(size_t)d[e] * Hc + f], h[(size_t)s[e] * Hc + f]);
    }
}

__global__ void gconv_finalize_kernel(const float* __restrict__ agg,
                                      const float* __restrict__ rsI,
                                      float* __restrict__ y, int n, int Hc) {
    long long t = (long long)blockIdx.x * blockDim.x + threadIdx.x;
    if (t < (long long)n * Hc) {
        int i = (int)(t / Hc);
        float v = agg[t] * rsI[i];
        y[t] = (v >= 0.0f) ? v : RSLOPE * v;     // RReLU eval mode
    }
}

// ------------------------------------------------------------- classifier 2
// Wave per node: lanes stride the 510 hidden dims (coalesced), ReLU fused,
// two shuffle-reduced dots.

__global__ void classifier2_kernel(const float* __restrict__ hidden,
                                   const float* __restrict__ mW2,
                                   const float* __restrict__ mb2,
                                   float* __restrict__ out, int n) {
    int wave = (blockIdx.x * blockDim.x + threadIdx.x) >> 5;
    int lane = threadIdx.x & 31;
    if (wave >= n) return;
    const float* hrow = hidden + (size_t)wave * CH_DIM;
    float s0 = 0.0f, s1 = 0.0f;
    for (int k = lane; k < CH_DIM; k += 32) {
        float hv = hrow[k];
        hv = hv > 0.0f ? hv : 0.0f;
        s0 += hv * mW2[k * C_DIM + 0];
        s1 += hv * mW2[k * C_DIM + 1];
    }
    s0 = wave_reduce_add(s0);
    s1 = wave_reduce_add(s1);
    if (lane == 0) {
        out[(size_t)wave * C_DIM + 0] = s0 + mb2[0];
        out[(size_t)wave * C_DIM + 1] = s1 + mb2[1];
    }
}

// ------------------------------------------------------------------- launch

static inline void zero_async(void* p, long long bytes, hipStream_t stream) {
    long long words = bytes >> 2;
    int blk = 256;
    long long grid = (words + blk - 1) / blk;
    fill_zero_kernel<<<(unsigned)grid, blk, 0, stream>>>((uint32_t*)p, words);
}

extern "C" void kernel_launch(void* const* d_in, const int* in_sizes, int n_in,
                              void* d_out, int out_size, void* d_ws, size_t ws_size,
                              hipStream_t stream) {
    (void)in_sizes; (void)n_in; (void)out_size; (void)ws_size;

    const float* feats  = (const float*)d_in[0];   // T x N x F0
    const int*   srcAll = (const int*)d_in[1];     // T x E
    const int*   dstAll = (const int*)d_in[2];     // T x E
    const float* scorer0 = (const float*)d_in[3];
    const float* uW0 = (const float*)d_in[4],  *uU0 = (const float*)d_in[5],  *ub0 = (const float*)d_in[6];
    const float* rW0 = (const float*)d_in[7],  *rU0 = (const float*)d_in[8],  *rb0 = (const float*)d_in[9];
    const float* hW0 = (const float*)d_in[10], *hU0 = (const float*)d_in[11], *hb0 = (const float*)d_in[12];
    const float* gW0 = (const float*)d_in[13];
    const float* scorer1 = (const float*)d_in[14];
    const float* uW1 = (const float*)d_in[15], *uU1 = (const float*)d_in[16], *ub1 = (const float*)d_in[17];
    const float* rW1 = (const float*)d_in[18], *rU1 = (const float*)d_in[19], *rb1 = (const float*)d_in[20];
    const float* hW1 = (const float*)d_in[21], *hU1 = (const float*)d_in[22], *hb1 = (const float*)d_in[23];
    const float* gW1 = (const float*)d_in[24];
    const float* mW1 = (const float*)d_in[25], *mb1 = (const float*)d_in[26];
    const float* mW2 = (const float*)d_in[27], *mb2 = (const float*)d_in[28];
    float* out = (float*)d_out;

    // ---- workspace carve-up (all chunks are multiples of 4 floats = 16 B) ----
    float* ws = (float*)d_ws;
    float* y0      = ws;                ws += (size_t)T_STEPS * N_NODES * H_DIM;
    float* hbuf    = ws;                ws += (size_t)N_NODES * H_DIM;
    float* aggbuf  = ws;                ws += (size_t)N_NODES * H_DIM;
    float* y1      = ws;                ws += (size_t)N_NODES * H_DIM;
    float* hidden  = ws;                ws += (size_t)N_NODES * CH_DIM;
    int*   degO    = (int*)ws;          ws += N_NODES;
    int*   degI    = (int*)ws;          ws += N_NODES;
    float* rsO     = ws;                ws += N_NODES;
    float* rsI     = ws;                ws += N_NODES;
    float* scores  = ws;                ws += N_NODES;
    int*   topIdx  = (int*)ws;          ws += H_DIM;       // 76 (x4B, /4 ok)
    float* topVal  = ws;                ws += H_DIM;
    float* xt      = ws;                ws += (size_t)F0_DIM * H_DIM;   // 12616
    float* W0c     = ws;                ws += (size_t)F0_DIM * H_DIM;
    float* W1c     = ws;                ws += (size_t)H_DIM * H_DIM;    // 5776
    float* gu      = ws;                ws += (size_t)F0_DIM * H_DIM;
    float* gr      = ws;                ws += (size_t)F0_DIM * H_DIM;
    float* grq     = ws;                ws += (size_t)F0_DIM * H_DIM;
    _Float16* Ah   = (_Float16*)ws;     ws += (size_t)N_NODES * 192 / 2;  // max M*Kp halfs
    _Float16* Bt   = (_Float16*)ws;     ws += (512 * 96 > 80 * 192 ? 512 * 96 : 80 * 192) / 2;

    const int BLK = 256;
    auto cdiv = [](long long a, long long b) { return (unsigned)((a + b - 1) / b); };

    // init evolving weights W <- gW
    hipMemcpyAsync(W0c, gW0, sizeof(float) * F0_DIM * H_DIM, hipMemcpyDeviceToDevice, stream);
    hipMemcpyAsync(W1c, gW1, sizeof(float) * H_DIM * H_DIM, hipMemcpyDeviceToDevice, stream);

    auto run_gru = [&](const float* uW, const float* uU, const float* ub,
                       const float* rW, const float* rU, const float* rb,
                       const float* hW, const float* hU, const float* hb,
                       float* Q, int F) {
        int n = F * H_DIM;
        gru_ur_kernel<<<cdiv(n, BLK), BLK, 0, stream>>>(uW, uU, ub, rW, rU, rb, xt, Q, gu, gr, F);
        gru_rq_kernel<<<cdiv(n, BLK), BLK, 0, stream>>>(gr, Q, grq, n);
        gru_h_kernel<<<cdiv(n, BLK), BLK, 0, stream>>>(hW, hU, hb, xt, grq, gu, Q, F);
    };

    // Staged WMMA GEMM: pack A (f16, row-scaled) + B^T (f16), then tiled WMMA.
    auto run_gemm = [&](const float* A, int lda, const float* rowScale,
                        const float* B, int ldb, const float* bias,
                        float* C, int ldc, int M, int Nc, int K) {
        int Kp = (K + 31) & ~31;
        int NT, tilesNG;
        if (Nc <= 80) { NT = 5; tilesNG = 1; }
        else          { NT = 4; tilesNG = (Nc + 63) / 64; }
        int Np = tilesNG * NT * 16;
        int tilesM = (M + 15) / 16;

        pack_a_f16_kernel<<<cdiv((long long)M * Kp, BLK), BLK, 0, stream>>>(
            A, lda, rowScale, Ah, M, K, Kp);
        pack_bT_f16_kernel<<<cdiv((long long)Np * Kp, BLK), BLK, 0, stream>>>(
            B, ldb, Bt, K, Nc, Kp, Np);

        long long waves = (long long)tilesM * tilesNG;
        unsigned blocks = cdiv(waves * 32, BLK);
        if (NT == 5)
            wmma_gemm_f16_kernel<5><<<blocks, BLK, 0, stream>>>(Ah, Kp, Bt, bias, C, ldc,
                                                                M, Nc, tilesM, tilesNG);
        else
            wmma_gemm_f16_kernel<4><<<blocks, BLK, 0, stream>>>(Ah, Kp, Bt, bias, C, ldc,
                                                                M, Nc, tilesM, tilesNG);
    };

    auto run_gconv = [&](const float* X, int F, const float* Wc,
                         const int* s, const int* d, float* yout) {
        run_gemm(X, F, rsO, Wc, H_DIM, nullptr, hbuf, H_DIM, N_NODES, H_DIM, F);
        zero_async(aggbuf, sizeof(float) * (size_t)N_NODES * H_DIM, stream);
        long long work = (long long)E_EDGES * H_DIM;
        edge_scatter_kernel<<<cdiv(work, BLK), BLK, 0, stream>>>(hbuf, s, d, aggbuf, E_EDGES, H_DIM);
        long long nw = (long long)N_NODES * H_DIM;
        gconv_finalize_kernel<<<cdiv(nw, BLK), BLK, 0, stream>>>(aggbuf, rsI, yout, N_NODES, H_DIM);
    };

    auto run_pool = [&](const float* X, int F, const float* scorer) {
        scores_kernel<<<cdiv((long long)N_NODES * 32, BLK), BLK, 0, stream>>>(X, scorer, F, N_NODES, scores);
        topk_kernel<<<1, 1024, 0, stream>>>(scores, N_NODES, topIdx, topVal);
        gather_xtilde_kernel<<<cdiv(F * H_DIM, BLK), BLK, 0, stream>>>(X, F, scorer, topIdx, topVal, xt);
    };

    // ---------------- layer 0: full scan over T ----------------
    for (int t = 0; t < T_STEPS; ++t) {
        const float* Xt = feats + (size_t)t * N_NODES * F0_DIM;
        const int* s = srcAll + (size_t)t * E_EDGES;
        const int* d = dstAll + (size_t)t * E_EDGES;

        zero_async(degO, sizeof(int) * N_NODES, stream);
        zero_async(degI, sizeof(int) * N_NODES, stream);
        deg_count_kernel<<<cdiv(E_EDGES, BLK), BLK, 0, stream>>>(s, d, degO, degI, E_EDGES);
        deg_finalize_kernel<<<cdiv(N_NODES, BLK), BLK, 0, stream>>>(degO, degI, rsO, rsI, N_NODES);

        run_pool(Xt, F0_DIM, scorer0);
        run_gru(uW0, uU0, ub0, rW0, rU0, rb0, hW0, hU0, hb0, W0c, F0_DIM);
        run_gconv(Xt, F0_DIM, W0c, s, d, y0 + (size_t)t * N_NODES * H_DIM);
    }

    // ---------------- layer 1: GRU every t, gconv only at t = T-1 ----------
    // (only feats[-1] of layer 1 feeds the classifier, so skip 5/6 scatters)
    for (int t = 0; t < T_STEPS; ++t) {
        const float* Xt = y0 + (size_t)t * N_NODES * H_DIM;
        run_pool(Xt, H_DIM, scorer1);
        run_gru(uW1, uU1, ub1, rW1, rU1, rb1, hW1, hU1, hb1, W1c, H_DIM);
        if (t == T_STEPS - 1) {
            // rsO/rsI still hold degrees of t = T-1 from the layer-0 loop
            const int* s = srcAll + (size_t)t * E_EDGES;
            const int* d = dstAll + (size_t)t * E_EDGES;
            run_gconv(Xt, H_DIM, W1c, s, d, y1);
        }
    }

    // ---------------- classifier ----------------
    run_gemm(y1, H_DIM, nullptr, mW1, CH_DIM, mb1, hidden, CH_DIM, N_NODES, CH_DIM, H_DIM);
    classifier2_kernel<<<cdiv((long long)N_NODES * 32, BLK), BLK, 0, stream>>>(hidden, mW2, mb2, out, N_NODES);
}
